// EncoderNSE_9053791060379
// MI455X (gfx1250) — compile-verified, matrix operand order checked by reading, not run on hardware
//
#include <hip/hip_runtime.h>
#include <hip/hip_bf16.h>
#include <stdint.h>

// Problem constants (D, T, B) = (512, 256, 64)
#define Dh 512
#define Th 256
#define Bh 64
#define G4 2048  // 4*D gate width

typedef __attribute__((ext_vector_type(16))) __bf16 v16bf;
typedef __attribute__((ext_vector_type(8)))  float  v8f;
typedef __attribute__((ext_vector_type(4)))  unsigned int u32x4;
typedef __attribute__((ext_vector_type(8)))  int i32x8;
typedef __attribute__((ext_vector_type(4)))  int i32x4;

__device__ __forceinline__ uint16_t f2bf(float x) {
    union { float f; unsigned u; } c; c.f = x;
    unsigned u = c.u;
    u += 0x7FFFu + ((u >> 16) & 1u);   // round-to-nearest-even
    return (uint16_t)(u >> 16);
}
__device__ __forceinline__ float sigf(float x) { return 1.0f / (1.0f + __expf(-x)); }

// ---------------------------------------------------------------------------
// Tensor Data Mover: async 2-D tile load Global -> LDS (CDNA5 ISA ch.8 D#).
// data_size = 2 bytes (bf16). LDS padding: 1 DWORD pad per 256 DWORDs so each
// 512-element row segment has a 257-DWORD stride (kills LDS bank conflicts).
// ---------------------------------------------------------------------------
__device__ __forceinline__ void tdm_load_2d_bf16(const void* gptr, unsigned ldsOff,
                                                 int rowElems, int rows)
{
    unsigned long long ga = (unsigned long long)(uintptr_t)gptr;
    // Group0: count=1 | lds_addr | global_addr[56:0] | type=2 (bits 127:126)
    u32x4 g0 = { 1u,
                 ldsOff,
                 (unsigned)ga,
                 ((unsigned)(ga >> 32) & 0x01FFFFFFu) | 0x80000000u };
    // Group1: data_size=1(2B) @16, pad_enable @20, pad_interval=7(256dw) @22,
    //         pad_amount=0(1dw) @25; tensor_dim0 @48; tensor_dim1 @80;
    //         tile_dim0 @112; tile_dim1 @128; tensor_dim0_stride @160.
    i32x8 g1 = { (int)0x01D10000u,
                 (int)(((unsigned)rowElems & 0xFFFFu) << 16),
                 (int)((((unsigned)rowElems >> 16) & 0xFFFFu) | ((unsigned)rows << 16)),
                 (int)(((unsigned)rowElems & 0xFFFFu) << 16),
                 rows,
                 rowElems,
                 0, 0 };
    i32x4 z4 = { 0, 0, 0, 0 };
#if __clang_major__ >= 23
    i32x8 z8 = { 0, 0, 0, 0, 0, 0, 0, 0 };
    __builtin_amdgcn_tensor_load_to_lds(g0, g1, z4, z4, z8, 0);
#else
    __builtin_amdgcn_tensor_load_to_lds(g0, g1, z4, z4, 0);
#endif
}

// ---------------------------------------------------------------------------
// bf16 WMMA GEMM: Out[64][N] = A[64][1024] * W^T + bias, A = [X | H] split at KX.
// A panel staged into LDS by the TDM (padded rows); B streamed from L2 with
// double-buffered fragments and two accumulators (pipelined K-loop).
// One 16x16 tile per wave, 8 waves/block -> block covers 64 rows x 32 cols.
// ---------------------------------------------------------------------------
template<int KX>
__global__ void EncNSE_gemm_wmma(const uint16_t* __restrict__ X,
                                 const uint16_t* __restrict__ H,
                                 const uint16_t* __restrict__ W,     // [N][1024] bf16 row-major
                                 const float*    __restrict__ bias,  // [N]
                                 float*          __restrict__ Out,   // [64][N] f32
                                 uint16_t*       __restrict__ OutBf, // optional bf16 copy
                                 int N)
{
    constexpr int K = 1024;
    extern __shared__ uint16_t As[];   // 128 padded segments * 257 dwords = 131584 B

    int tid  = threadIdx.x;
    int lane = tid & 31;
    int wave = (blockIdx.x * (int)blockDim.x + tid) >> 5;
    int tileM = wave & 3;
    int tileN = wave >> 2;
    int half  = lane >> 4;
    int l15   = lane & 15;
    int mrow  = tileM * 16 + l15;
    int ncol  = tileN * 16 + l15;

    // --- TDM: stage A panel(s) into LDS (issued by wave 0 only) ---
    if (tid == 0) {
        unsigned base = (unsigned)(uintptr_t)As;   // flat LDS addr low 32 bits = LDS offset
        tdm_load_2d_bf16(X, base, KX, Bh);
        if constexpr (KX < K)
            tdm_load_2d_bf16(H, base + 64u * 257u * 4u, K - KX, Bh);
        __builtin_amdgcn_s_wait_tensorcnt(0);
    }

    const unsigned* Wrow32 = (const unsigned*)(W + (size_t)ncol * K);
    const unsigned* AsU32  = (const unsigned*)As;

    union FragU { unsigned u[8]; v16bf v; };
    auto loadB = [&](int kb, FragU& dst) {
        const unsigned* B32 = Wrow32 + (kb << 4);
        #pragma unroll
        for (int v = 0; v < 8; ++v)
            dst.u[v] = B32[(v & 3) + 4 * half + ((v >= 4) ? 8 : 0)];
    };
    auto loadA = [&](int kb, FragU& dst) {
        int kbase = kb * 32;
        const unsigned* A32;
        if constexpr (KX == K) {
            int kw = kbase >> 1;                        // 0..511 dwords in row
            A32 = AsU32 + ((mrow << 1) + (kw >> 8)) * 257 + (kw & 255);
        } else {                                        // KX == 512
            if (kbase < KX) A32 = AsU32 + mrow * 257 + (kbase >> 1);
            else            A32 = AsU32 + (64 + mrow) * 257 + ((kbase - KX) >> 1);
        }
        #pragma unroll
        for (int v = 0; v < 8; ++v)
            dst.u[v] = A32[(v & 3) + 4 * half + ((v >= 4) ? 8 : 0)];
    };

    FragU b0, b1, a0, a1;
    loadB(0, b0);                                       // overlap with TDM
    __builtin_prefetch((const void*)(Wrow32 + 256), 0, 0);
    __syncthreads();                                    // LDS A panel ready

    v8f acc0 = {0.f,0.f,0.f,0.f,0.f,0.f,0.f,0.f};
    v8f acc1 = {0.f,0.f,0.f,0.f,0.f,0.f,0.f,0.f};
    #pragma unroll
    for (int kb = 0; kb < K / 32; kb += 2) {
        loadB(kb + 1, b1);
        loadA(kb, a0);
        acc0 = __builtin_amdgcn_wmma_f32_16x16x32_bf16(false, a0.v, false, b0.v,
                                                       (short)0, acc0, false, false);
        if (kb + 2 < K / 32) loadB(kb + 2, b0);
        loadA(kb + 1, a1);
        acc1 = __builtin_amdgcn_wmma_f32_16x16x32_bf16(false, a1.v, false, b1.v,
                                                       (short)0, acc1, false, false);
    }

    float bv = bias[ncol];
    #pragma unroll
    for (int r = 0; r < 8; ++r) {
        int m = tileM * 16 + r + 8 * half;   // C/D layout: VGPR r holds M=r (+8 hi lanes)
        float val = acc0[r] + acc1[r] + bv;
        Out[(size_t)m * N + ncol] = val;
        if (OutBf) OutBf[(size_t)m * N + ncol] = f2bf(val);
    }
}

// ---------------------------------------------------------------------------
// Read-LSTM activation + attention softmax + composition input build (1 block / batch)
// ---------------------------------------------------------------------------
__global__ void EncNSE_read_attn(const float* __restrict__ G,
                                 float* __restrict__ c_r, float* __restrict__ h_r,
                                 uint16_t* __restrict__ h_r_bf,
                                 const float* __restrict__ mem,
                                 float* __restrict__ zbuf,
                                 uint16_t* __restrict__ comp_bf)
{
    __shared__ float hs[Dh];
    __shared__ float zl[Th];
    __shared__ float red[Th];
    int b   = blockIdx.x;
    int tid = threadIdx.x;
    const float* Gb = G + (size_t)b * G4;

    for (int d = tid; d < Dh; d += blockDim.x) {
        float gi = Gb[d], gf = Gb[Dh + d], gg = Gb[2 * Dh + d], go = Gb[3 * Dh + d];
        float cn = sigf(gf) * c_r[b * Dh + d] + sigf(gi) * tanhf(gg);
        float hn = sigf(go) * tanhf(cn);
        c_r[b * Dh + d] = cn;
        h_r[b * Dh + d] = hn;
        h_r_bf[b * Dh + d] = f2bf(hn);
        hs[d] = hn;
    }
    __syncthreads();

    const float* memb = mem + (size_t)b * Dh * Th;
    float s = 0.f;                                   // score for slot t' = tid
    for (int d = 0; d < Dh; ++d) s = fmaf(hs[d], memb[d * Th + tid], s);

    red[tid] = s; __syncthreads();
    for (int off = Th / 2; off > 0; off >>= 1) {
        if (tid < off) red[tid] = fmaxf(red[tid], red[tid + off]);
        __syncthreads();
    }
    float mx = red[0]; __syncthreads();
    float e = __expf(s - mx);
    red[tid] = e; __syncthreads();
    for (int off = Th / 2; off > 0; off >>= 1) {
        if (tid < off) red[tid] += red[tid + off];
        __syncthreads();
    }
    float z = e / red[0];
    zl[tid] = z;
    zbuf[b * Th + tid] = z;
    __syncthreads();

    for (int d = tid; d < Dh; d += blockDim.x) {
        const float* row = memb + (size_t)d * Th;
        float acc = 0.f;
        #pragma unroll 4
        for (int tt = 0; tt < Th; ++tt) acc = fmaf(row[tt], zl[tt], acc);
        comp_bf[b * 2 * Dh + 2 * d]     = f2bf(hs[d]);  // interleave: stack([h_r, m], -1)
        comp_bf[b * 2 * Dh + 2 * d + 1] = f2bf(acc);
    }
}

// ---------------------------------------------------------------------------
// Fused write-LSTM activation + memory update + outs + snapshots.
// 1 block per (b,d): thread 0 does the gate nonlinearity, broadcasts h_w via
// LDS; 256 threads update the mem row over t'.
// ---------------------------------------------------------------------------
__global__ void EncNSE_write_mem(const float* __restrict__ G,
                                 float* __restrict__ c_w, uint16_t* __restrict__ h_w_bf,
                                 const float* __restrict__ h_r, const float* __restrict__ c_r,
                                 const float* __restrict__ zbuf, const int* __restrict__ lens,
                                 int t,
                                 float* __restrict__ mem, float* __restrict__ outs,
                                 float* __restrict__ shw, float* __restrict__ scw,
                                 float* __restrict__ shr, float* __restrict__ scr,
                                 float* __restrict__ smem)
{
    __shared__ float sh;
    int bd = blockIdx.x;
    int tt = threadIdx.x;
    int b  = bd >> 9;
    int d  = bd & (Dh - 1);
    bool sel = (t == lens[b] - 1);
    if (tt == 0) {
        const float* Gb = G + (size_t)b * G4;
        float gi = Gb[d], gf = Gb[Dh + d], gg = Gb[2 * Dh + d], go = Gb[3 * Dh + d];
        float cn = sigf(gf) * c_w[bd] + sigf(gi) * tanhf(gg);
        float hn = sigf(go) * tanhf(cn);
        c_w[bd] = cn;
        h_w_bf[bd] = f2bf(hn);
        outs[(size_t)t * Bh * Dh + bd] = hn;
        if (sel) { shw[bd] = hn; scw[bd] = cn; shr[bd] = h_r[bd]; scr[bd] = c_r[bd]; }
        sh = hn;
    }
    __syncthreads();
    float hw = sh;
    float z  = zbuf[b * Th + tt];
    size_t idx = (size_t)bd * Th + tt;
    float old = mem[idx];
    float nv = fmaf(hw - old, z, old);   // mem*(1-z) + h_w*z
    mem[idx] = nv;
    if (sel) smem[idx] = nv;
}

// ---------------------------------------------------------------------------
// Setup: weight concat+bf16, Wc bf16, embs transpose + bf16 copy
// ---------------------------------------------------------------------------
__global__ void EncNSE_prep_cat(const float* __restrict__ Wih, const float* __restrict__ Whh,
                                const float* __restrict__ bih, const float* __restrict__ bhh,
                                uint16_t* __restrict__ Wcat, float* __restrict__ bias)
{
    int idx = blockIdx.x * blockDim.x + threadIdx.x;  // 2048*1024
    int n = idx >> 10;
    int k = idx & 1023;
    float v = (k < Dh) ? Wih[n * Dh + k] : Whh[n * Dh + (k - Dh)];
    Wcat[idx] = f2bf(v);
    if (k == 0) bias[n] = bih[n] + bhh[n];
}

__global__ void EncNSE_prep_wc(const float* __restrict__ Wc, uint16_t* __restrict__ WcBf)
{
    int idx = blockIdx.x * blockDim.x + threadIdx.x;  // 512*1024
    WcBf[idx] = f2bf(Wc[idx]);
}

__global__ void EncNSE_transpose(const float* __restrict__ embs, float* __restrict__ mem,
                                 uint16_t* __restrict__ xbf)
{
    __shared__ float tile[32][33];
    int b  = blockIdx.z;
    int t0 = blockIdx.x * 32, d0 = blockIdx.y * 32;
    int tx = threadIdx.x, ty = threadIdx.y;  // 32 x 8
    #pragma unroll
    for (int i = 0; i < 4; ++i) {
        int tt = t0 + ty + 8 * i;
        size_t src = (size_t)tt * Bh * Dh + (size_t)b * Dh + d0 + tx;
        float v = embs[src];
        tile[ty + 8 * i][tx] = v;
        xbf[src] = f2bf(v);
    }
    __syncthreads();
    #pragma unroll
    for (int i = 0; i < 4; ++i) {
        int d = d0 + ty + 8 * i;
        mem[(size_t)b * Dh * Th + (size_t)d * Th + t0 + tx] = tile[tx][ty + 8 * i];
    }
}

// ---------------------------------------------------------------------------
extern "C" void kernel_launch(void* const* d_in, const int* in_sizes, int n_in,
                              void* d_out, int out_size, void* d_ws, size_t ws_size,
                              hipStream_t stream)
{
    (void)in_sizes; (void)n_in; (void)out_size; (void)ws_size;
    const float* embs  = (const float*)d_in[0];
    const int*   lens  = (const int*)  d_in[1];
    const float* Wih_r = (const float*)d_in[2];
    const float* Whh_r = (const float*)d_in[3];
    const float* bih_r = (const float*)d_in[4];
    const float* bhh_r = (const float*)d_in[5];
    const float* Wih_w = (const float*)d_in[6];
    const float* Whh_w = (const float*)d_in[7];
    const float* bih_w = (const float*)d_in[8];
    const float* bhh_w = (const float*)d_in[9];
    const float* Wc    = (const float*)d_in[10];
    const float* bc    = (const float*)d_in[11];

    // ---- workspace carve-up (~61 MB) ----
    char* wsp = (char*)d_ws;
    auto carve = [&](size_t bytes) { char* p = wsp; wsp += (bytes + 255) & ~(size_t)255; return p; };
    float*    mem     = (float*)   carve((size_t)Bh * Dh * Th * 4);   // 33.5 MB (L2-resident)
    uint16_t* xbf     = (uint16_t*)carve((size_t)Th * Bh * Dh * 2);   // 16.8 MB
    uint16_t* WcatR   = (uint16_t*)carve((size_t)G4 * 1024 * 2);
    uint16_t* WcatW   = (uint16_t*)carve((size_t)G4 * 1024 * 2);
    uint16_t* WcBf    = (uint16_t*)carve((size_t)Dh * 1024 * 2);
    float*    biasR   = (float*)   carve((size_t)G4 * 4);
    float*    biasW   = (float*)   carve((size_t)G4 * 4);
    float*    states  = (float*)   carve((size_t)4 * Bh * Dh * 4);
    float *h_r = states, *c_r = states + Bh * Dh, *c_w = states + 2 * Bh * Dh;
    uint16_t* h_r_bf  = (uint16_t*)carve((size_t)Bh * Dh * 2);
    uint16_t* h_w_bf  = (uint16_t*)carve((size_t)Bh * Dh * 2);
    uint16_t* c_bf    = (uint16_t*)carve((size_t)Bh * Dh * 2);
    uint16_t* comp_bf = (uint16_t*)carve((size_t)Bh * 1024 * 2);
    float*    Gbuf    = (float*)   carve((size_t)Bh * G4 * 4);
    float*    cbuf    = (float*)   carve((size_t)Bh * Dh * 4);
    float*    zbuf    = (float*)   carve((size_t)Bh * Th * 4);

    // ---- output layout (flat, reference return order) ----
    float* out  = (float*)d_out;
    float* outs = out;                                  // (T,B,D)
    float* shw  = outs + (size_t)Th * Bh * Dh;          // (1,B,D)
    float* scw  = shw + (size_t)Bh * Dh;                // (1,B,D)
    float* smem = scw + (size_t)Bh * Dh;                // (B,D,T)
    float* shr  = smem + (size_t)Bh * Dh * Th;          // (1,B,D)
    float* scr  = shr + (size_t)Bh * Dh;                // (1,B,D)

    // ---- one-time setup ----
    hipMemsetAsync(states, 0, (size_t)4 * Bh * Dh * 4, stream);
    hipMemsetAsync(h_r_bf, 0, (size_t)Bh * Dh * 2, stream);
    hipMemsetAsync(h_w_bf, 0, (size_t)Bh * Dh * 2, stream);
    EncNSE_prep_cat<<<(G4 * 1024) / 256, 256, 0, stream>>>(Wih_r, Whh_r, bih_r, bhh_r, WcatR, biasR);
    EncNSE_prep_cat<<<(G4 * 1024) / 256, 256, 0, stream>>>(Wih_w, Whh_w, bih_w, bhh_w, WcatW, biasW);
    EncNSE_prep_wc<<<(Dh * 1024) / 256, 256, 0, stream>>>(Wc, WcBf);
    {
        dim3 tg(Th / 32, Dh / 32, Bh), tb(32, 8);
        EncNSE_transpose<<<tg, tb, 0, stream>>>(embs, mem, xbf);
    }

    // ---- sequential scan: 4 kernels per step, stream order == data deps ----
    const int    gatesBlocks = (4 * (G4 / 16)) / 8;     // 64 blocks
    const int    wcBlocks    = (4 * (Dh / 16)) / 8;     // 16 blocks
    const size_t ashm        = 128u * 257u * 4u;        // padded A panel in LDS (131584 B)
    for (int t = 0; t < Th; ++t) {
        const uint16_t* xt = xbf + (size_t)t * Bh * Dh;
        // read-LSTM gates: [x_t | h_r] @ [Wih_r;Whh_r]^T
        EncNSE_gemm_wmma<512><<<gatesBlocks, 256, ashm, stream>>>(xt, h_r_bf, WcatR, biasR,
                                                                  Gbuf, (uint16_t*)nullptr, G4);
        EncNSE_read_attn<<<Bh, Th, 0, stream>>>(Gbuf, c_r, h_r, h_r_bf, mem, zbuf, comp_bf);
        // composition: interleave(h_r, m) @ Wc^T + bc  (bf16 result feeds next GEMM)
        EncNSE_gemm_wmma<1024><<<wcBlocks, 256, ashm, stream>>>(comp_bf, (const uint16_t*)nullptr,
                                                                WcBf, bc, cbuf, c_bf, Dh);
        // write-LSTM gates: [c | h_w] @ [Wih_w;Whh_w]^T
        EncNSE_gemm_wmma<512><<<gatesBlocks, 256, ashm, stream>>>(c_bf, h_w_bf, WcatW, biasW,
                                                                  Gbuf, (uint16_t*)nullptr, G4);
        // fused write-LSTM activation + rank-1 memory update + outputs/snapshots
        EncNSE_write_mem<<<Bh * Dh, Th, 0, stream>>>(Gbuf, c_w, h_w_bf, h_r, c_r, zbuf, lens, t,
                                                     mem, outs, shw, scw, shr, scr, smem);
    }
}
